// BioEncoder_30167850287707
// MI455X (gfx1250) — matrix operand order repaired and checked by reading, not compiled, not run on hardware
//
#include <hip/hip_runtime.h>
#include <hip/hip_bf16.h>
#include <math.h>

#define N_NODES 200000
#define N_EDGES 800000
#define N_GRAPHS 4096
#define DIM_DRUG 78
#define DIM_CELL 954
#define HIDN 128
#define BN_EPS 1e-5f

typedef __bf16 bf16;
typedef __bf16 v16bf __attribute__((ext_vector_type(16)));
typedef float  v8f   __attribute__((ext_vector_type(8)));
typedef int    v4i   __attribute__((vector_size(16)));

#if __has_builtin(__builtin_amdgcn_global_load_async_to_lds_b128)
#define USE_ASYNC_LDS 1
#else
#define USE_ASYNC_LDS 0
#endif

__device__ __forceinline__ void wait_async_zero() {
#if __has_builtin(__builtin_amdgcn_s_wait_asynccnt)
  __builtin_amdgcn_s_wait_asynccnt(0);
#elif USE_ASYNC_LDS
  asm volatile("s_wait_asynccnt 0x0" ::: "memory");
#endif
}

__device__ __forceinline__ void sched_fence() {
#if __has_builtin(__builtin_amdgcn_sched_barrier)
  __builtin_amdgcn_sched_barrier(0);
#endif
}

// Copy 32 bytes/thread of the W fragment chunk into LDS (8 KB per chunk,
// 256 threads). Async global->LDS (ASYNCcnt) when available.
__device__ __forceinline__ void stage_chunk(const bf16* __restrict__ gsrc,
                                            bf16* ldst, int tid) {
  const char* g = (const char*)gsrc + (size_t)tid * 32;
  char* l = (char*)ldst + (size_t)tid * 32;
#if USE_ASYNC_LDS
  __builtin_amdgcn_global_load_async_to_lds_b128(
      (__attribute__((address_space(1))) v4i*)(g),
      (__attribute__((address_space(3))) v4i*)(l), 0, 0);
  __builtin_amdgcn_global_load_async_to_lds_b128(
      (__attribute__((address_space(1))) v4i*)(g + 16),
      (__attribute__((address_space(3))) v4i*)(l + 16), 0, 0);
#else
  ((uint4*)l)[0] = ((const uint4*)g)[0];
  ((uint4*)l)[1] = ((const uint4*)g)[1];
#endif
}

// ---------------------------------------------------------------------------
// Fragment pack kernels (CDNA5 wave32 WMMA bf16 layouts):
// A 16x32: lane holds row M=lane%16; elem j -> K = (j/8)*16 + (lane>=16)*8 + (j%8)
// B 32x16: lane holds col N=lane%16; elem j -> K = (j/8)*16 + (lane>=16)*8 + (j%8)
// ---------------------------------------------------------------------------
__global__ void pack_a_kernel(const float* __restrict__ X, bf16* __restrict__ out,
                              int Mtiles, int KC, int K, int ldx) {
  size_t tid = (size_t)blockIdx.x * blockDim.x + threadIdx.x;
  size_t total = (size_t)Mtiles * KC * 512;
  if (tid >= total) return;
  int j    = (int)(tid & 15);
  int lane = (int)((tid >> 4) & 31);
  size_t rest = tid >> 9;
  int kc = (int)(rest % KC);
  int mt = (int)(rest / KC);
  int row = mt * 16 + (lane & 15);
  int k = kc * 32 + ((j >> 3) << 4) + ((lane >> 4) << 3) + (j & 7);
  float v = (k < K) ? X[(size_t)row * ldx + k] : 0.0f;
  out[tid] = (bf16)v;
}

__global__ void pack_b_kernel(const float* __restrict__ W, bf16* __restrict__ out,
                              int KC, int K) {
  int tid = blockIdx.x * blockDim.x + threadIdx.x;
  int total = KC * 8 * 512;
  if (tid >= total) return;
  int j    = tid & 15;
  int lane = (tid >> 4) & 31;
  int nt   = (tid >> 9) & 7;
  int kc   = tid >> 12;
  int col = nt * 16 + (lane & 15);
  int k = kc * 32 + ((j >> 3) << 4) + ((lane >> 4) << 3) + (j & 7);
  float v = (k < K) ? W[(size_t)k * HIDN + col] : 0.0f;
  out[tid] = (bf16)v;
}

// ---------------------------------------------------------------------------
// WMMA GEMM: 8 waves/block; each wave computes a 16(M) x 128(N) stripe with
// 8 f32 accumulators. W chunks staged in double-buffered LDS via async
// global->LDS, B fragments batch-loaded with ds_load_b128, then 8
// back-to-back WMMAs per K-chunk (sched_barrier keeps the phases separate).
// Apack: [Mtiles][KC][32 lanes][16 bf16], Wpack: [KC][8 ntiles][32][16 bf16]
// ---------------------------------------------------------------------------
__global__ void __launch_bounds__(256) gemm_wmma_kernel(
    const bf16* __restrict__ Apack, const bf16* __restrict__ Wpack,
    const float* __restrict__ bias, float* __restrict__ out,
    int Mtiles, int KC, int relu_out) {
  __shared__ __align__(32) bf16 sw[2][8 * 512];  // 2 x 8KB W chunks
  int tid = threadIdx.x;
  int wave = tid >> 5;
  int lane = tid & 31;
  int mtile = blockIdx.x * 8 + wave;
  bool active = mtile < Mtiles;  // wave-uniform

  v8f acc[8];
#pragma unroll
  for (int nt = 0; nt < 8; ++nt) acc[nt] = 0.0f;

  const bf16* aptr = Apack + ((size_t)mtile * KC) * 512 + (size_t)lane * 16;

  // prime buffer 0
  stage_chunk(Wpack, sw[0], tid);
  wait_async_zero();
  __syncthreads();

  for (int kc = 0; kc < KC; ++kc) {
    int cur = kc & 1;
    // overlap: stage next chunk into the other buffer while computing
    if (kc + 1 < KC) stage_chunk(Wpack + (size_t)(kc + 1) * 4096, sw[1 - cur], tid);

    v16bf a = {};
    if (active) a = *(const v16bf*)(aptr + (size_t)kc * 512);

    v16bf b[8];
    const bf16* bp = sw[cur] + (size_t)lane * 16;
#pragma unroll
    for (int nt = 0; nt < 8; ++nt) b[nt] = *(const v16bf*)(bp + (size_t)nt * 512);

    sched_fence();  // batch all ds_loads before the WMMA chain

#pragma unroll
    for (int nt = 0; nt < 8; ++nt)
      acc[nt] = __builtin_amdgcn_wmma_f32_16x16x32_bf16(
          false, a, false, b[nt], (short)0, acc[nt], false, false);

    sched_fence();

    wait_async_zero();
    __syncthreads();
  }

  if (!active) return;
  // D layout: VGPR r holds M = r (lanes 0-15) or 8+r (lanes 16-31), N = lane%16
  int col0 = lane & 15;
  int rowbase = mtile * 16 + ((lane >> 4) << 3);
#pragma unroll
  for (int nt = 0; nt < 8; ++nt) {
    int col = nt * 16 + col0;
    float bv = bias ? bias[col] : 0.0f;
#pragma unroll
    for (int r = 0; r < 8; ++r) {
      float v = acc[nt][r] + bv;
      if (relu_out) v = fmaxf(v, 0.0f);
      out[(size_t)(rowbase + r) * HIDN + col] = v;
    }
  }
}

// ---------------------------------------------------------------------------
// GCN normalization + scatter
// ---------------------------------------------------------------------------
__global__ void deg_init_kernel(float* deg, int n) {
  int i = blockIdx.x * blockDim.x + threadIdx.x;
  if (i < n) deg[i] = 1.0f;  // self-loop
}
__global__ void deg_edges_kernel(float* deg, const long long* __restrict__ ei, int ne) {
  int e = blockIdx.x * blockDim.x + threadIdx.x;
  if (e < ne) atomicAdd(&deg[ei[(size_t)ne + e]], 1.0f);
}
__global__ void deg_finish_kernel(float* deg, int n) {
  int i = blockIdx.x * blockDim.x + threadIdx.x;
  if (i < n) deg[i] = rsqrtf(fmaxf(deg[i], 1e-12f));
}

// out[i,c] = bias[c] + H[i,c] * dinv[i]^2   (self-loop contribution + bias)
__global__ void scat_init_kernel(const float* __restrict__ H, const float* __restrict__ dinv,
                                 const float* __restrict__ bias, float* __restrict__ out, int n) {
  size_t t = (size_t)blockIdx.x * blockDim.x + threadIdx.x;
  if (t >= (size_t)n * HIDN) return;
  int i = (int)(t >> 7);
  int c = (int)(t & 127);
  float di = dinv[i];
  out[t] = bias[c] + H[t] * di * di;
}

// out[dst] += H[src] * dinv[src]*dinv[dst]  (4 channels per thread)
__global__ void scat_edges_kernel(const float* __restrict__ H, const float* __restrict__ dinv,
                                  const long long* __restrict__ ei, float* __restrict__ out, int ne) {
  size_t t = (size_t)blockIdx.x * blockDim.x + threadIdx.x;
  if (t >= (size_t)ne * 32) return;
  int e = (int)(t >> 5);
  int q = (int)((t & 31) << 2);
  long long s = ei[e];
  long long d = ei[(size_t)ne + e];
  float w = dinv[s] * dinv[d];
  const float4 h = *(const float4*)(H + (size_t)s * HIDN + q);
  float* o = out + (size_t)d * HIDN + q;
  atomicAdd(o + 0, h.x * w);
  atomicAdd(o + 1, h.y * w);
  atomicAdd(o + 2, h.z * w);
  atomicAdd(o + 3, h.w * w);
}

// ---------------------------------------------------------------------------
// BatchNorm (training stats), fused pooling, activations
// ---------------------------------------------------------------------------
__global__ void zero_stats_kernel(float* stats) {
  if (threadIdx.x < 256) stats[threadIdx.x] = 0.0f;
}

__global__ void bn_stats_kernel(const float* __restrict__ x, float* __restrict__ stats,
                                int M, int relu_in) {
  int c = threadIdx.x;  // 128 channels
  float s = 0.0f, s2 = 0.0f;
  for (int i = blockIdx.x; i < M; i += gridDim.x) {
    float v = x[(size_t)i * HIDN + c];
    if (relu_in) v = fmaxf(v, 0.0f);
    s += v;
    s2 += v * v;
  }
  atomicAdd(&stats[c], s);
  atomicAdd(&stats[HIDN + c], s2);
}

__global__ void bn_apply_kernel(float* __restrict__ x, const float* __restrict__ stats,
                                const float* __restrict__ gamma, const float* __restrict__ beta,
                                int M, int relu_in) {
  size_t t = (size_t)blockIdx.x * blockDim.x + threadIdx.x;
  if (t >= (size_t)M * HIDN) return;
  int c = (int)(t & 127);
  float mu = stats[c] / (float)M;
  float var = stats[HIDN + c] / (float)M - mu * mu;
  float v = x[t];
  if (relu_in) v = fmaxf(v, 0.0f);
  x[t] = gamma[c] * (v - mu) * rsqrtf(var + BN_EPS) + beta[c];
}

__global__ void seg_init_kernel(float* out, int n) {
  int i = blockIdx.x * blockDim.x + threadIdx.x;
  if (i < n) out[i] = -INFINITY;
}

// fused: y = BN(relu(x)); out[ibatch[i],c] = max(out, y)
__global__ void bn_segmax_kernel(const float* __restrict__ x, const float* __restrict__ stats,
                                 const float* __restrict__ gamma, const float* __restrict__ beta,
                                 const long long* __restrict__ ibatch, float* __restrict__ out, int M) {
  size_t t = (size_t)blockIdx.x * blockDim.x + threadIdx.x;
  if (t >= (size_t)M * HIDN) return;
  int i = (int)(t >> 7);
  int c = (int)(t & 127);
  float mu = stats[c] / (float)M;
  float var = stats[HIDN + c] / (float)M - mu * mu;
  float v = fmaxf(x[t], 0.0f);
  float z = gamma[c] * (v - mu) * rsqrtf(var + BN_EPS) + beta[c];
  atomicMax(out + (size_t)ibatch[i] * HIDN + c, z);
}

__global__ void tanh_kernel(float* x, int n) {
  int i = blockIdx.x * blockDim.x + threadIdx.x;
  if (i < n) x[i] = tanhf(x[i]);
}

// ---------------------------------------------------------------------------
static inline unsigned cdiv(size_t a, size_t b) { return (unsigned)((a + b - 1) / b); }

extern "C" void kernel_launch(void* const* d_in, const int* in_sizes, int n_in,
                              void* d_out, int out_size, void* d_ws, size_t ws_size,
                              hipStream_t stream) {
  const float*     drug   = (const float*)d_in[0];
  const long long* ei     = (const long long*)d_in[1];
  const long long* ibatch = (const long long*)d_in[2];
  const float*     gexpr  = (const float*)d_in[3];
  const float *W1 = (const float*)d_in[4],  *b1  = (const float*)d_in[5];
  const float *g1 = (const float*)d_in[6],  *be1 = (const float*)d_in[7];
  const float *W2 = (const float*)d_in[8],  *b2  = (const float*)d_in[9];
  const float *g2 = (const float*)d_in[10], *be2 = (const float*)d_in[11];
  const float *Wc1 = (const float*)d_in[12], *bc1  = (const float*)d_in[13];
  const float *gc1 = (const float*)d_in[14], *bec1 = (const float*)d_in[15];
  const float *Wc2 = (const float*)d_in[16], *bc2  = (const float*)d_in[17];
  float* out = (float*)d_out;
  (void)in_sizes; (void)n_in; (void)out_size; (void)ws_size;

  // workspace layout (all 256B-aligned)
  char* ws = (char*)d_ws;
  float* A     = (float*)(ws + 0);          // 200000*128*4 = 102,400,000
  float* B     = (float*)(ws + 102400000);  // 102,400,000
  float* dinv  = (float*)(ws + 204800000);  // 800,000
  float* stats = (float*)(ws + 205600000);  // 1,024
  bf16*  Apack = (bf16*) (ws + 205601024);  // up to 12500*4*1024 = 51,200,000
  bf16*  Wpack = (bf16*) (ws + 256801024);  // up to 30*8*1024 = 245,760
  float* C1    = (float*)(ws + 257046784);  // 4096*128*4 = 2,097,152

  const int MT_D = N_NODES / 16;   // 12500
  const int MT_C = N_GRAPHS / 16;  // 256
  const int KC1 = (DIM_DRUG + 31) / 32;   // 3
  const int KC2 = HIDN / 32;              // 4
  const int KCC = (DIM_CELL + 31) / 32;   // 30

  // ---- GCN degree norm ----
  deg_init_kernel<<<cdiv(N_NODES, 256), 256, 0, stream>>>(dinv, N_NODES);
  deg_edges_kernel<<<cdiv(N_EDGES, 256), 256, 0, stream>>>(dinv, ei, N_EDGES);
  deg_finish_kernel<<<cdiv(N_NODES, 256), 256, 0, stream>>>(dinv, N_NODES);

  // ---- drug layer 1: H1 = drug @ W1 ----
  pack_b_kernel<<<cdiv((size_t)KC1 * 8 * 512, 256), 256, 0, stream>>>(W1, Wpack, KC1, DIM_DRUG);
  pack_a_kernel<<<cdiv((size_t)MT_D * KC1 * 512, 256), 256, 0, stream>>>(drug, Apack, MT_D, KC1, DIM_DRUG, DIM_DRUG);
  gemm_wmma_kernel<<<cdiv(MT_D, 8), 256, 0, stream>>>(Apack, Wpack, nullptr, A, MT_D, KC1, 0);
  scat_init_kernel<<<cdiv((size_t)N_NODES * HIDN, 256), 256, 0, stream>>>(A, dinv, b1, B, N_NODES);
  scat_edges_kernel<<<cdiv((size_t)N_EDGES * 32, 256), 256, 0, stream>>>(A, dinv, ei, B, N_EDGES);
  zero_stats_kernel<<<1, 256, 0, stream>>>(stats);
  bn_stats_kernel<<<512, 128, 0, stream>>>(B, stats, N_NODES, 1);
  bn_apply_kernel<<<cdiv((size_t)N_NODES * HIDN, 256), 256, 0, stream>>>(B, stats, g1, be1, N_NODES, 1);

  // ---- drug layer 2: H2 = X1 @ W2 ----
  pack_b_kernel<<<cdiv((size_t)KC2 * 8 * 512, 256), 256, 0, stream>>>(W2, Wpack, KC2, HIDN);
  pack_a_kernel<<<cdiv((size_t)MT_D * KC2 * 512, 256), 256, 0, stream>>>(B, Apack, MT_D, KC2, HIDN, HIDN);
  gemm_wmma_kernel<<<cdiv(MT_D, 8), 256, 0, stream>>>(Apack, Wpack, nullptr, A, MT_D, KC2, 0);
  scat_init_kernel<<<cdiv((size_t)N_NODES * HIDN, 256), 256, 0, stream>>>(A, dinv, b2, B, N_NODES);
  scat_edges_kernel<<<cdiv((size_t)N_EDGES * 32, 256), 256, 0, stream>>>(A, dinv, ei, B, N_EDGES);
  zero_stats_kernel<<<1, 256, 0, stream>>>(stats);
  bn_stats_kernel<<<512, 128, 0, stream>>>(B, stats, N_NODES, 1);
  seg_init_kernel<<<cdiv((size_t)N_GRAPHS * HIDN, 256), 256, 0, stream>>>(out, N_GRAPHS * HIDN);
  bn_segmax_kernel<<<cdiv((size_t)N_NODES * HIDN, 256), 256, 0, stream>>>(B, stats, g2, be2, ibatch, out, N_NODES);

  // ---- cell branch ----
  pack_b_kernel<<<cdiv((size_t)KCC * 8 * 512, 256), 256, 0, stream>>>(Wc1, Wpack, KCC, DIM_CELL);
  pack_a_kernel<<<cdiv((size_t)MT_C * KCC * 512, 256), 256, 0, stream>>>(gexpr, Apack, MT_C, KCC, DIM_CELL, DIM_CELL);
  gemm_wmma_kernel<<<cdiv(MT_C, 8), 256, 0, stream>>>(Apack, Wpack, bc1, C1, MT_C, KCC, 0);
  tanh_kernel<<<cdiv((size_t)N_GRAPHS * HIDN, 256), 256, 0, stream>>>(C1, N_GRAPHS * HIDN);
  zero_stats_kernel<<<1, 256, 0, stream>>>(stats);
  bn_stats_kernel<<<64, 128, 0, stream>>>(C1, stats, N_GRAPHS, 0);
  bn_apply_kernel<<<cdiv((size_t)N_GRAPHS * HIDN, 256), 256, 0, stream>>>(C1, stats, gc1, bec1, N_GRAPHS, 0);
  pack_b_kernel<<<cdiv((size_t)KC2 * 8 * 512, 256), 256, 0, stream>>>(Wc2, Wpack, KC2, HIDN);
  pack_a_kernel<<<cdiv((size_t)MT_C * KC2 * 512, 256), 256, 0, stream>>>(C1, Apack, MT_C, KC2, HIDN, HIDN);
  gemm_wmma_kernel<<<cdiv(MT_C, 8), 256, 0, stream>>>(Apack, Wpack, bc2, out + (size_t)N_GRAPHS * HIDN, MT_C, KC2, 1);
}
